// GMNLayerWithNorm_5317169513089
// MI455X (gfx1250) — matrix operand order repaired and checked by reading, not compiled, not run on hardware
//
#include <hip/hip_runtime.h>
#include <hip/hip_bf16.h>

#define NN 65536
#define DD 32
#define HH 4
#define HDIM 8
#define BB 64
#define NSS 512
#define EE 1048576

typedef __attribute__((ext_vector_type(16))) _Float16 v16h;
typedef __attribute__((ext_vector_type(8)))  float    v8f;

#define AS_GLOBAL __attribute__((address_space(1)))
#define AS_LDS    __attribute__((address_space(3)))

#if __has_builtin(__builtin_amdgcn_global_load_async_to_lds_b32) && \
    __has_builtin(__builtin_amdgcn_s_wait_asynccnt)
#define HAVE_ASYNC_LDS 1
#else
#define HAVE_ASYNC_LDS 0
#endif

// ---------------- WMMA fragment helpers (wave32, 16x16x32 f16) ----------------
// A (16x32 f16): lanes 0-15 -> M=lane, K 0..15;
//                lanes 16-31 -> M=lane-16, K 16..31
__device__ inline v16h load_a_f16(const float* base, int ld) {
    int l  = threadIdx.x & 31;
    int m  = l & 15;
    int ko = (l < 16) ? 0 : 16;
    v16h a;
#pragma unroll
    for (int i = 0; i < 8; ++i) {
        int k = ko + 2 * i;
        a[2 * i]     = (_Float16)base[m * ld + k];
        a[2 * i + 1] = (_Float16)base[m * ld + k + 1];
    }
    return a;
}

// B (32x16 f16): lanes 0-15 -> N=lane, K = 2v,2v+1; lanes 16-31 -> N=lane-16, K = 16+2v,16+2v+1
// transpose=false: B[k][n] = W[k*32 + n]     (y = x @ W)
// transpose=true : B[k][n] = W[n*32 + k]     (y = x @ W^T)
__device__ inline v16h load_b_f16(const float* W, int ncol0, bool transpose) {
    int l  = threadIdx.x & 31;
    int n  = (l & 15) + ncol0;
    int kb = (l < 16) ? 0 : 16;
    v16h b;
#pragma unroll
    for (int v = 0; v < 8; ++v) {
        int k0 = kb + 2 * v;
        float e0 = transpose ? W[n * DD + k0]       : W[k0 * DD + n];
        float e1 = transpose ? W[n * DD + k0 + 1]   : W[(k0 + 1) * DD + n];
        b[2 * v]     = (_Float16)e0;
        b[2 * v + 1] = (_Float16)e1;
    }
    return b;
}

// C/D (16x16 f32): lane l -> n = l&15; VGPR r -> m = r + 8*(l>=16)
__device__ inline void store_c(float* dst, int ncol0, const v8f& c) {
    int l  = threadIdx.x & 31;
    int n  = (l & 15) + ncol0;
    int mb = (l < 16) ? 0 : 8;
#pragma unroll
    for (int r = 0; r < 8; ++r) dst[(mb + r) * DD + n] = c[r];
}

__device__ inline void store_c_bias(float* dst, int ncol0, const v8f& c, const float* bias) {
    int l  = threadIdx.x & 31;
    int n  = (l & 15) + ncol0;
    int mb = (l < 16) ? 0 : 8;
    float bv = bias[n];
#pragma unroll
    for (int r = 0; r < 8; ++r) dst[(mb + r) * DD + n] = c[r] + bv;
}

__device__ inline v8f wmma16(v16h a, v16h b, v8f c) {
    return __builtin_amdgcn_wmma_f32_16x16x32_f16(false, a, false, b, (short)0, c, false, false);
}

// ---------------- K1: xp = x @ gat_w  (8 waves/block, one 16-row tile per wave) ----------------
__global__ void __launch_bounds__(256) k_gat_proj(const float* __restrict__ x,
                                                  const float* __restrict__ W,
                                                  float* __restrict__ xp) {
    int tile = blockIdx.x * 8 + (threadIdx.x >> 5);
    const float* src = x + (size_t)tile * 16 * DD;
    v16h a  = load_a_f16(src, DD);
    v16h b0 = load_b_f16(W, 0,  false);
    v16h b1 = load_b_f16(W, 16, false);
    v8f c0 = {}; v8f c1 = {};
    c0 = wmma16(a, b0, c0);
    c1 = wmma16(a, b1, c1);
    float* dst = xp + (size_t)tile * 16 * DD;
    store_c(dst, 0,  c0);
    store_c(dst, 16, c1);
}

// ---------------- K2: per (node, head) attention scores + init ----------------
__global__ void k_att_scores(const float* __restrict__ xp,
                             const float* __restrict__ att_src,
                             const float* __restrict__ att_dst,
                             float* __restrict__ asrc, float* __restrict__ adst,
                             float* __restrict__ mmax, float* __restrict__ ssum,
                             float* __restrict__ acc) {
    int t = blockIdx.x * blockDim.x + threadIdx.x;
    if (t >= NN * HH) return;
    int n = t >> 2, h = t & 3;
    float as = 0.f, ad = 0.f;
#pragma unroll
    for (int j = 0; j < HDIM; ++j) {
        float v = xp[n * DD + h * HDIM + j];
        as += v * att_src[h * HDIM + j];
        ad += v * att_dst[h * HDIM + j];
        acc[n * DD + h * HDIM + j] = 0.f;
    }
    asrc[t] = as;
    adst[t] = ad;
    mmax[t] = -3.0e38f;
    ssum[t] = 0.f;
}

// ---------------- float atomic max (sign-aware int/uint ordering) ----------------
__device__ inline void atomicMaxF(float* addr, float val) {
    if (val >= 0.f) atomicMax((int*)addr, __float_as_int(val));
    else            atomicMin((unsigned int*)addr, __float_as_uint(val));
}

// ---------------- K3a: segment max over edges (+ self loops) ----------------
__global__ void k_edge_max(const long long* __restrict__ ei,
                           const float* __restrict__ asrc,
                           const float* __restrict__ adst,
                           float* __restrict__ mmax) {
    long long t = (long long)blockIdx.x * blockDim.x + threadIdx.x;
    if (t >= (long long)EE + NN) return;
    int s, d;
    if (t < EE) { s = (int)ei[t]; d = (int)ei[(long long)EE + t]; }
    else        { s = d = (int)(t - EE); }
#pragma unroll
    for (int h = 0; h < HH; ++h) {
        float al = asrc[s * HH + h] + adst[d * HH + h];
        al = (al > 0.f) ? al : 0.2f * al;       // leaky_relu(0.2)
        atomicMaxF(&mmax[d * HH + h], al);
    }
}

// ---------------- K3b: exp-weights, segment sum + weighted feature scatter ----------------
__global__ void k_edge_acc(const long long* __restrict__ ei,
                           const float* __restrict__ asrc,
                           const float* __restrict__ adst,
                           const float* __restrict__ mmax,
                           const float* __restrict__ xp,
                           float* __restrict__ ssum,
                           float* __restrict__ acc) {
    long long t = (long long)blockIdx.x * blockDim.x + threadIdx.x;
    if (t >= (long long)EE + NN) return;
    int s, d;
    if (t < EE) { s = (int)ei[t]; d = (int)ei[(long long)EE + t]; }
    else        { s = d = (int)(t - EE); }
#pragma unroll
    for (int h = 0; h < HH; ++h) {
        float al = asrc[s * HH + h] + adst[d * HH + h];
        al = (al > 0.f) ? al : 0.2f * al;
        float e = __expf(al - mmax[d * HH + h]);
        atomicAdd(&ssum[d * HH + h], e);
#pragma unroll
        for (int j = 0; j < HDIM; ++j)
            atomicAdd(&acc[d * DD + h * HDIM + j], e * xp[s * DD + h * HDIM + j]);
    }
}

// ---------------- K4: h1 = LayerNorm(x + gat_out + bias), wave per node ----------------
__global__ void __launch_bounds__(128) k_h1(const float* __restrict__ x,
                                            const float* __restrict__ acc,
                                            const float* __restrict__ ssum,
                                            const float* __restrict__ gat_bias,
                                            const float* __restrict__ g,
                                            const float* __restrict__ b,
                                            float* __restrict__ h1) {
    int nid = blockIdx.x * 4 + (threadIdx.x >> 5);
    int c   = threadIdx.x & 31;
    float val = x[(size_t)nid * DD + c]
              + acc[(size_t)nid * DD + c] / (ssum[nid * HH + (c >> 3)] + 1e-16f)
              + gat_bias[c];
    float sum = val;
#pragma unroll
    for (int off = 16; off > 0; off >>= 1) sum += __shfl_xor(sum, off, 32);
    float mu = sum * (1.0f / DD);
    float dv = val - mu;
    float sq = dv * dv;
#pragma unroll
    for (int off = 16; off > 0; off >>= 1) sq += __shfl_xor(sq, off, 32);
    float var = sq * (1.0f / DD);
    h1[(size_t)nid * DD + c] = dv * rsqrtf(var + 1e-5f) * g[c] + b[c];
}

// ---------------- K5: Q/K/V projections (h1 @ W^T + b), 8 waves/block ----------------
__global__ void __launch_bounds__(256) k_qkv(const float* __restrict__ h1,
                                             const float* __restrict__ wq, const float* __restrict__ bq,
                                             const float* __restrict__ wk, const float* __restrict__ bk,
                                             const float* __restrict__ wv, const float* __restrict__ bv,
                                             float* __restrict__ qp, float* __restrict__ kp,
                                             float* __restrict__ vp) {
    int tile = blockIdx.x * 8 + (threadIdx.x >> 5);
    const float* src = h1 + (size_t)tile * 16 * DD;
    v16h a = load_a_f16(src, DD);

    {   v16h b0 = load_b_f16(wq, 0, true), b1 = load_b_f16(wq, 16, true);
        v8f c0 = {}, c1 = {};
        c0 = wmma16(a, b0, c0); c1 = wmma16(a, b1, c1);
        float* dst = qp + (size_t)tile * 16 * DD;
        store_c_bias(dst, 0, c0, bq); store_c_bias(dst, 16, c1, bq); }
    {   v16h b0 = load_b_f16(wk, 0, true), b1 = load_b_f16(wk, 16, true);
        v8f c0 = {}, c1 = {};
        c0 = wmma16(a, b0, c0); c1 = wmma16(a, b1, c1);
        float* dst = kp + (size_t)tile * 16 * DD;
        store_c_bias(dst, 0, c0, bk); store_c_bias(dst, 16, c1, bk); }
    {   v16h b0 = load_b_f16(wv, 0, true), b1 = load_b_f16(wv, 16, true);
        v8f c0 = {}, c1 = {};
        c0 = wmma16(a, b0, c0); c1 = wmma16(a, b1, c1);
        float* dst = vp + (size_t)tile * 16 * DD;
        store_c_bias(dst, 0, c0, bv); store_c_bias(dst, 16, c1, bv); }
}

// ---------------- K6: cross-attention, online softmax. block = (pair, dir), 512 thr ----------------
__global__ void __launch_bounds__(512) k_attn(const float* __restrict__ qp,
                                              const float* __restrict__ kp,
                                              const float* __restrict__ vp,
                                              const long long* __restrict__ idx0,
                                              const long long* __restrict__ idx1,
                                              float* __restrict__ ao) {
    int pair = blockIdx.x >> 1;
    int dir  = blockIdx.x & 1;
    const long long* qi = (dir == 0) ? idx0 : idx1;
    const long long* ki = (dir == 0) ? idx1 : idx0;
    int t  = threadIdx.x;                       // one query per thread
    int qn = (int)qi[pair * NSS + t];

    float qv[DD];
#pragma unroll
    for (int c = 0; c < DD; ++c) qv[c] = qp[(size_t)qn * DD + c];

    float m4[HH], l4[HH], o[DD];
#pragma unroll
    for (int h = 0; h < HH; ++h) { m4[h] = -3.0e38f; l4[h] = 0.f; }
#pragma unroll
    for (int c = 0; c < DD; ++c) o[c] = 0.f;

    __shared__ float kS[64 * DD];
    __shared__ float vS[64 * DD];

    const float scale = 0.3535533906f;          // 1/sqrt(8)

    for (int chunk = 0; chunk < NSS / 64; ++chunk) {
        __syncthreads();
        for (int i = t; i < 64 * DD; i += 512) {
            int j = i >> 5, c = i & 31;
            int kn = (int)ki[pair * NSS + chunk * 64 + j];
#if HAVE_ASYNC_LDS
            // async DMA to LDS, tracked by ASYNCcnt (global_load_async_to_lds_b32)
            __builtin_amdgcn_global_load_async_to_lds_b32(
                (AS_GLOBAL int*)(kp + (size_t)kn * DD + c),
                (AS_LDS int*)(&kS[i]), 0, 0);
            __builtin_amdgcn_global_load_async_to_lds_b32(
                (AS_GLOBAL int*)(vp + (size_t)kn * DD + c),
                (AS_LDS int*)(&vS[i]), 0, 0);
#else
            kS[i] = kp[(size_t)kn * DD + c];
            vS[i] = vp[(size_t)kn * DD + c];
#endif
        }
        // prefetch next chunk's K rows into cache (global_prefetch_b8)
        if (chunk + 1 < NSS / 64 && t < 64) {
            int kn = (int)ki[pair * NSS + (chunk + 1) * 64 + t];
            __builtin_prefetch(&kp[(size_t)kn * DD], 0, 1);
            __builtin_prefetch(&vp[(size_t)kn * DD], 0, 1);
        }
#if HAVE_ASYNC_LDS
        __builtin_amdgcn_s_wait_asynccnt(0);
#endif
        __syncthreads();
#pragma unroll 4
        for (int j = 0; j < 64; ++j) {
#pragma unroll
            for (int h = 0; h < HH; ++h) {
                float sc = 0.f;
#pragma unroll
                for (int dch = 0; dch < HDIM; ++dch)
                    sc += qv[h * HDIM + dch] * kS[j * DD + h * HDIM + dch];
                sc *= scale;
                float nm   = fmaxf(m4[h], sc);
                float corr = __expf(m4[h] - nm);
                float p    = __expf(sc - nm);
                l4[h] = l4[h] * corr + p;
#pragma unroll
                for (int dch = 0; dch < HDIM; ++dch)
                    o[h * HDIM + dch] = o[h * HDIM + dch] * corr + p * vS[j * DD + h * HDIM + dch];
                m4[h] = nm;
            }
        }
    }
#pragma unroll
    for (int c = 0; c < DD; ++c)
        ao[(size_t)qn * DD + c] = o[c] / (l4[c >> 3] + 1e-30f);
}

// ---------------- K7: h2 = ao @ wo^T + bo; out = LayerNorm(h1 + h2), 8 waves/block ----------------
__global__ void __launch_bounds__(256) k_out_ln(const float* __restrict__ ao,
                                                const float* __restrict__ wo,
                                                const float* __restrict__ bo,
                                                const float* __restrict__ h1,
                                                const float* __restrict__ g,
                                                const float* __restrict__ b,
                                                float* __restrict__ out) {
    int tile = blockIdx.x * 8 + (threadIdx.x >> 5);
    const float* src = ao + (size_t)tile * 16 * DD;
    v16h a  = load_a_f16(src, DD);
    v16h b0 = load_b_f16(wo, 0,  true);
    v16h b1 = load_b_f16(wo, 16, true);
    v8f c0 = {}, c1 = {};
    c0 = wmma16(a, b0, c0);
    c1 = wmma16(a, b1, c1);

    int l  = threadIdx.x & 31;
    int n  = l & 15;
    int mb = (l < 16) ? 0 : 8;
    float bo0 = bo[n], bo1 = bo[n + 16];
    float g0 = g[n], g1 = g[n + 16], bb0 = b[n], bb1 = b[n + 16];

#pragma unroll
    for (int r = 0; r < 8; ++r) {
        int row = tile * 16 + mb + r;
        float v0 = c0[r] + bo0 + h1[(size_t)row * DD + n];
        float v1 = c1[r] + bo1 + h1[(size_t)row * DD + 16 + n];
        // row sums: each 16-lane half holds one row's 16+16 columns
        float sum = v0 + v1;
#pragma unroll
        for (int msk = 1; msk < 16; msk <<= 1) sum += __shfl_xor(sum, msk, 32);
        float mu = sum * (1.0f / DD);
        float d0 = v0 - mu, d1 = v1 - mu;
        float sq = d0 * d0 + d1 * d1;
#pragma unroll
        for (int msk = 1; msk < 16; msk <<= 1) sq += __shfl_xor(sq, msk, 32);
        float inv = rsqrtf(sq * (1.0f / DD) + 1e-5f);
        out[(size_t)row * DD + n]      = d0 * inv * g0 + bb0;
        out[(size_t)row * DD + 16 + n] = d1 * inv * g1 + bb1;
    }
}

// ---------------- launcher ----------------
extern "C" void kernel_launch(void* const* d_in, const int* in_sizes, int n_in,
                              void* d_out, int out_size, void* d_ws, size_t ws_size,
                              hipStream_t stream) {
    const float*      x        = (const float*)d_in[0];
    const long long*  ei       = (const long long*)d_in[1];
    const long long*  idx0     = (const long long*)d_in[2];
    const long long*  idx1     = (const long long*)d_in[3];
    const float*      gat_w    = (const float*)d_in[4];
    const float*      att_src  = (const float*)d_in[5];
    const float*      att_dst  = (const float*)d_in[6];
    const float*      gat_bias = (const float*)d_in[7];
    const float*      ln1_g    = (const float*)d_in[8];
    const float*      ln1_b    = (const float*)d_in[9];
    const float*      wq       = (const float*)d_in[10];
    const float*      wk       = (const float*)d_in[11];
    const float*      wv       = (const float*)d_in[12];
    const float*      bq       = (const float*)d_in[13];
    const float*      bk       = (const float*)d_in[14];
    const float*      bv       = (const float*)d_in[15];
    const float*      wo       = (const float*)d_in[16];
    const float*      bo       = (const float*)d_in[17];
    const float*      ln2_g    = (const float*)d_in[18];
    const float*      ln2_b    = (const float*)d_in[19];
    float*            out      = (float*)d_out;

    float* ws   = (float*)d_ws;
    float* xp   = ws;                       // NN*DD
    float* asrc = xp   + (size_t)NN * DD;   // NN*HH
    float* adst = asrc + (size_t)NN * HH;   // NN*HH
    float* mmax = adst + (size_t)NN * HH;   // NN*HH
    float* ssum = mmax + (size_t)NN * HH;   // NN*HH
    float* acc  = ssum + (size_t)NN * HH;   // NN*DD
    float* h1   = acc  + (size_t)NN * DD;   // NN*DD
    float* qp   = h1   + (size_t)NN * DD;   // NN*DD
    float* kp   = qp   + (size_t)NN * DD;   // NN*DD
    float* vp   = kp   + (size_t)NN * DD;   // NN*DD
    float* ao   = vp   + (size_t)NN * DD;   // NN*DD

    const int tiles = NN / 16;              // 4096

    k_gat_proj<<<tiles / 8, 256, 0, stream>>>(x, gat_w, xp);
    k_att_scores<<<(NN * HH + 255) / 256, 256, 0, stream>>>(
        xp, att_src, att_dst, asrc, adst, mmax, ssum, acc);
    {
        long long tot = (long long)EE + NN;
        int blocks = (int)((tot + 255) / 256);
        k_edge_max<<<blocks, 256, 0, stream>>>(ei, asrc, adst, mmax);
        k_edge_acc<<<blocks, 256, 0, stream>>>(ei, asrc, adst, mmax, xp, ssum, acc);
    }
    k_h1<<<NN / 4, 128, 0, stream>>>(x, acc, ssum, gat_bias, ln1_g, ln1_b, h1);
    k_qkv<<<tiles / 8, 256, 0, stream>>>(h1, wq, bq, wk, bk, wv, bv, qp, kp, vp);
    k_attn<<<BB * 2, 512, 0, stream>>>(qp, kp, vp, idx0, idx1, ao);
    k_out_ln<<<tiles / 8, 256, 0, stream>>>(ao, wo, bo, h1, ln2_g, ln2_b, out);
}
